// TransformerDecoderLayer_987842478858
// MI455X (gfx1250) — matrix operand order repaired and checked by reading, not compiled, not run on hardware
//
#include <hip/hip_runtime.h>
#include <hip/hip_bf16.h>
#include <stdint.h>

// ---------------- problem constants ----------------
#define DD   256
#define FFN  1024
#define NHD  8
#define DHD  32
#define BZc  8
#define TLc  1024
#define HWc  4096           // 64*64, L=1
#define NT   (BZc*TLc)      // 8192 target tokens
#define NS   (BZc*HWc)      // 32768 source tokens

// ---------------- CDNA5 WMMA types ----------------
typedef __attribute__((ext_vector_type(16))) __bf16 v16bf;
typedef __attribute__((ext_vector_type(8)))  float  v8f;
typedef __attribute__((ext_vector_type(4)))  unsigned int u32x4;

#if defined(__AMDGCN__) && __has_builtin(__builtin_amdgcn_tensor_load_to_lds)
#define HAVE_TDM 1
#else
#define HAVE_TDM 0
#endif

union Frag { v16bf v; u32x4 q[2]; };   // 32B: one 16x32 bf16 A/B fragment per lane

__device__ inline v8f vzero8() {
  v8f z;
#pragma unroll
  for (int i = 0; i < 8; ++i) z[i] = 0.0f;
  return z;
}

__device__ inline v8f bfwmma(v16bf a, v16bf b, v8f c) {
  // D = A(16x32 bf16) * B(32x16 bf16) + C(16x16 f32)
  return __builtin_amdgcn_wmma_f32_16x16x32_bf16(
      /*neg_a=*/false, a, /*neg_b=*/false, b,
      /*c_mod=*/(short)0, c, /*reuse_a=*/false, /*reuse_b=*/false);
}

__device__ inline __bf16 f2bf(float f) {
  union { float f; uint32_t u; } in; in.f = f;
  uint32_t u = in.u;
  uint32_t r = (u + 0x7FFFu + ((u >> 16) & 1u)) >> 16;   // round-to-nearest-even
  union { uint16_t s; __bf16 b; } out; out.s = (uint16_t)r;
  return out.b;
}

// ---- CDNA5 async global->LDS copy (ASYNCcnt path, ISA 15.18.3 op 98) ----
__device__ inline void async_copy_b128(uint32_t lds_off, const void* gaddr) {
  asm volatile("global_load_async_to_lds_b128 %0, %1, off"
               :: "v"(lds_off), "v"((uint64_t)(uintptr_t)gaddr) : "memory");
}
__device__ inline void wait_async_le1() {
  asm volatile("s_wait_asynccnt 0x1" ::: "memory");
}
__device__ inline void wait_async_le0() {
  asm volatile("s_wait_asynccnt 0x0" ::: "memory");
}

#if HAVE_TDM
// ---- CDNA5 Tensor Data Mover: 2D tile global->LDS (TENSORcnt path) ----
// D# per ISA ch.8: group0 = {count, lds_addr, global_addr[56:0], type=2},
// group1 = {data_size=2B, tensor_dim0/1, tile_dim0/1, tensor_dim0_stride}.
// This toolchain (clang-23 / amdgpu-toolchain) uses the 6-arg builtin form.
typedef __attribute__((ext_vector_type(8))) int i32x8;
typedef __attribute__((ext_vector_type(4))) int i32x4;
__device__ inline void tdm_load_2d(uint32_t lds_addr, const void* gaddr,
                                   uint32_t rowElems, uint32_t rows,
                                   uint32_t strideElems) {
  const uint64_t ga = (uint64_t)(uintptr_t)gaddr;
  u32x4 g0;
  g0[0] = 1u;                                                 // count=1 (user D#)
  g0[1] = lds_addr;                                           // LDS byte address
  g0[2] = (uint32_t)ga;                                       // global_addr[31:0]
  g0[3] = (uint32_t)((ga >> 32) & 0x01FFFFFFu) | (2u << 30);  // addr[56:32] | type=2
  i32x8 g1;
  g1[0] = (int)(1u << 16);                                    // data_size=1 -> 2 bytes
  g1[1] = (int)((rowElems & 0xFFFFu) << 16);                  // tensor_dim0[15:0]
  g1[2] = (int)(((rowElems >> 16) & 0xFFFFu) |
                ((rows & 0xFFFFu) << 16));                    // tdim0 hi | tensor_dim1 lo
  g1[3] = (int)(((rows >> 16) & 0xFFFFu) |
                ((rowElems & 0xFFFFu) << 16));                // tdim1 hi | tile_dim0
  g1[4] = (int)(rows & 0xFFFFu);                              // tile_dim1 (tile_dim2=0)
  g1[5] = (int)strideElems;                                   // tensor_dim0_stride[31:0]
  g1[6] = 0;                                                  // stride hi | dim1_stride lo
  g1[7] = 0;
  const i32x4 gz4 = {0, 0, 0, 0};
  const i32x8 gz8 = {0, 0, 0, 0, 0, 0, 0, 0};
  __builtin_amdgcn_tensor_load_to_lds(g0, g1, gz4, gz4, gz8, 0);
}
#endif

// ---------------- elementwise prep kernels ----------------
__global__ __launch_bounds__(256) void k_cvt_bf16(const float* __restrict__ s,
                                                  __bf16* __restrict__ d, int n) {
  int i = blockIdx.x * 256 + threadIdx.x;
  if (i < n) d[i] = f2bf(s[i]);
}

__global__ __launch_bounds__(256) void k_prep_tgt(const float* __restrict__ tgt,
                                                  const float* __restrict__ pos,
                                                  __bf16* __restrict__ qk,
                                                  __bf16* __restrict__ tb, int n) {
  int i = blockIdx.x * 256 + threadIdx.x;
  if (i < n) { float t = tgt[i]; qk[i] = f2bf(t + pos[i]); tb[i] = f2bf(t); }
}

// srcs (BZ, 1, D, H, W) -> s_bf [(b*HW+p)*D + c]
__global__ __launch_bounds__(256) void k_prep_srcs(const float* __restrict__ srcs,
                                                   __bf16* __restrict__ sb) {
  size_t i = (size_t)blockIdx.x * 256 + threadIdx.x;     // NS*DD total
  int c = (int)(i % DD);
  size_t sp = i / DD;
  int p = (int)(sp % HWc);
  int b = (int)(sp / HWc);
  sb[i] = f2bf(srcs[((size_t)b * DD + c) * HWc + p]);
}

// v_sa [b,t, h*32+d] -> vT_self [((b*NH+h)*32+d)*TL + t]
__global__ __launch_bounds__(256) void k_transpose_vself(const __bf16* __restrict__ v,
                                                         __bf16* __restrict__ vt) {
  size_t i = (size_t)blockIdx.x * 256 + threadIdx.x;     // NT*DD total
  int t = (int)(i % TLc);
  size_t r = i / TLc;
  int d = (int)(r % DHD); r /= DHD;
  int h = (int)(r % NHD);
  int b = (int)(r / NHD);
  vt[i] = v[((size_t)(b * TLc + t)) * DD + h * DHD + d];
}

// value [b,s,c] -> vT_cross [(b*D+c)*HW + s]
__global__ __launch_bounds__(256) void k_transpose_vcross(const __bf16* __restrict__ v,
                                                          __bf16* __restrict__ vt) {
  size_t i = (size_t)blockIdx.x * 256 + threadIdx.x;     // BZ*DD*HW total
  int s = (int)(i % HWc);
  size_t r = i / HWc;
  int c = (int)(r % DD);
  int b = (int)(r / DD);
  vt[i] = v[((size_t)(b * HWc + s)) * DD + c];
}

// ---------------- WMMA GEMM: C[M,N] = A[M,K] @ W[N,K]^T + bias ----------------
// 128-thread block (4 waves) computes a 32(M) x 256(N) tile.
// A tile (32x32 bf16, 2KB) staged via async global->LDS (ASYNCcnt), ping-pong
// LDS buffers; B chunks ping-pong between two register banks (no copies).
// Steady-state body is branch-free; final chunk pair is peeled.
// Requires K % 64 == 0 (K = 256 or 1024 here).
template <bool RELU, bool WF32, bool WBF16>
__global__ __launch_bounds__(128) void k_gemm(const __bf16* __restrict__ A,
                                              const __bf16* __restrict__ W,
                                              const float* __restrict__ bias,
                                              float* __restrict__ Cf,
                                              __bf16* __restrict__ Cb,
                                              int M, int N, int K) {
  __shared__ __bf16 sA[2][32 * 32];          // [buf][row][k-chunk of 32]

  const int tid  = threadIdx.x;
  const int lane = tid & 31;
  const int wv   = tid >> 5;                 // wave 0..3 -> N strip
  const int ln   = lane & 15;
  const int hi   = lane >> 4;
  const int kb   = hi ? 8 : 0;               // ISA 16-bit A-fragment K base
  const int mt   = blockIdx.x;               // M/32
  const int nb   = blockIdx.y;               // N/256
  const int n0   = nb * 256 + wv * 64;

  // async A-tile staging: thread copies one 16B segment (row=tid/4, seg=tid%4)
  const int arow = tid >> 2;
  const int aseg = tid & 3;
  const __bf16* Abase = A + ((size_t)(mt * 32 + arow)) * K + aseg * 8;
  const uint32_t ldsSeg = (uint32_t)(arow * 64 + aseg * 16);
  const uint32_t ldsA0 = (uint32_t)(uintptr_t)(&sA[0][0]) + ldsSeg;
  const uint32_t ldsA1 = (uint32_t)(uintptr_t)(&sA[1][0]) + ldsSeg;

  v8f acc[2][4];
#pragma unroll
  for (int t = 0; t < 2; ++t)
#pragma unroll
    for (int j = 0; j < 4; ++j) acc[t][j] = vzero8();

  Frag bA[4], bB[4];

  auto loadB = [&](Frag* bq, int k0) {
#pragma unroll
    for (int j = 0; j < 4; ++j) {
      const __bf16* wr = W + (size_t)(n0 + j * 16 + ln) * K + k0 + (hi ? 8 : 0);
      bq[j].q[0] = *(const u32x4*)(wr);
      bq[j].q[1] = *(const u32x4*)(wr + 16);
    }
  };
  auto mmaStep = [&](const __bf16* sa, Frag* bq) {
    Frag a0, a1;
    const __bf16* r0 = sa + ln * 32 + kb;          // row ln of 32-row tile
    a0.q[0] = *(const u32x4*)(r0);
    a0.q[1] = *(const u32x4*)(r0 + 16);
    const __bf16* r1 = sa + (16 + ln) * 32 + kb;   // row 16+ln
    a1.q[0] = *(const u32x4*)(r1);
    a1.q[1] = *(const u32x4*)(r1 + 16);
#pragma unroll
    for (int j = 0; j < 4; ++j) {
      acc[0][j] = bfwmma(a0.v, bq[j].v, acc[0][j]);
      acc[1][j] = bfwmma(a1.v, bq[j].v, acc[1][j]);
    }
  };

  // ---- pipeline prologue: chunk 0 ----
  async_copy_b128(ldsA0, Abase);
  loadB(bA, 0);

  // ---- steady state: branch-free, two chunks per trip ----
  int k0 = 0;
  for (; k0 + 64 < K; k0 += 64) {
    async_copy_b128(ldsA1, Abase + k0 + 32);   // stage chunk k0+32
    wait_async_le1();                          // chunk k0 resident in sA[0]
    __syncthreads();
    loadB(bB, k0 + 32);                        // overlap with WMMAs below
    mmaStep(&sA[0][0], bA);
    __syncthreads();                           // sA[0] consumed

    async_copy_b128(ldsA0, Abase + k0 + 64);   // stage chunk k0+64
    wait_async_le1();                          // chunk k0+32 resident in sA[1]
    __syncthreads();
    loadB(bA, k0 + 64);
    mmaStep(&sA[1][0], bB);
    __syncthreads();                           // sA[1] consumed
  }

  // ---- epilogue: final two chunks (k0, k0+32) ----
  async_copy_b128(ldsA1, Abase + k0 + 32);
  wait_async_le1();
  __syncthreads();
  loadB(bB, k0 + 32);
  mmaStep(&sA[0][0], bA);
  __syncthreads();

  wait_async_le0();
  __syncthreads();
  mmaStep(&sA[1][0], bB);

#pragma unroll
  for (int t = 0; t < 2; ++t) {
#pragma unroll
    for (int j = 0; j < 4; ++j) {
      const int n = n0 + j * 16 + ln;
      const float bs = bias ? bias[n] : 0.0f;
#pragma unroll
      for (int e = 0; e < 8; ++e) {
        const int mm = mt * 32 + t * 16 + e + 8 * hi;    // C layout: 8 rows / lane
        float v = acc[t][j][e] + bs;
        if (RELU) v = fmaxf(v, 0.0f);
        if (WF32)  Cf[(size_t)mm * N + n] = v;
        if (WBF16) Cb[(size_t)mm * N + n] = f2bf(v);
      }
    }
  }
}

// ---------------- flash attention (one wave per 16-query tile) ----------------
// Q,K: [*, rowstride=DD] bf16 (+headOff per head); VT: [HD][NK] bf16 (pre-transposed)
// K-tiles (32 keys x HD dims) are staged into LDS by the Tensor Data Mover
// (TENSORcnt), double-buffered; falls back to ASYNCcnt copies if TDM builtin
// is unavailable (e.g. host pass).
template <int HD>
__global__ __launch_bounds__(32) void k_flash(const __bf16* __restrict__ Q,
                                              const __bf16* __restrict__ K,
                                              const __bf16* __restrict__ VT,
                                              __bf16* __restrict__ O,
                                              int NK, float scale,
                                              size_t qBatch, size_t kBatch,
                                              size_t vtBatch, size_t vtHead,
                                              int headOff, int vtStride) {
  constexpr int NF = HD / 32;     // A fragments along head dim
  constexpr int ND = HD / 16;     // output d-tiles
  __shared__ __bf16 sK[2][32 * HD];   // staged key tiles [key][d]
  __shared__ float  sS[16 * 32];
  __shared__ __bf16 sP[16 * 32];
  __shared__ float  sL[16], sA[16];

  const int lane = threadIdx.x;
  const int qt = blockIdx.x, h = blockIdx.y, b = blockIdx.z;
  const __bf16* Qh = Q + (size_t)b * qBatch + (size_t)h * headOff;
  const __bf16* Kh = K + (size_t)b * kBatch + (size_t)h * headOff;
  const __bf16* Vh = VT + (size_t)b * vtBatch + (size_t)h * vtHead;
  __bf16* Oh = O + (size_t)b * qBatch + (size_t)h * headOff;

  const int ln = lane & 15;
  const int hi = lane >> 4;
  const int kb = hi ? 8 : 0;

  const uint32_t ldsK0 = (uint32_t)(uintptr_t)(&sK[0][0]);
  const uint32_t ldsK1 = (uint32_t)(uintptr_t)(&sK[1][0]);

  auto stageK = [&](int bufi, int kk) {
#if HAVE_TDM
    tdm_load_2d(bufi ? ldsK1 : ldsK0, Kh + (size_t)kk * DD, HD, 32, DD);
#else
    const uint32_t base = bufi ? ldsK1 : ldsK0;
    constexpr int SPR = HD / 8;        // 16B segments per key row
#pragma unroll
    for (int s = 0; s < HD / 8; ++s) { // (32 rows * SPR) / 32 lanes
      const int seg = s * 32 + lane;
      const int row = seg / SPR, col = seg % SPR;
      async_copy_b128(base + (uint32_t)seg * 16,
                      Kh + (size_t)(kk + row) * DD + col * 8);
    }
#endif
  };
  auto waitK1 = [&]() {
#if HAVE_TDM
    __builtin_amdgcn_s_wait_tensorcnt((short)1);
#else
    if constexpr (HD == 32) asm volatile("s_wait_asynccnt 0x4" ::: "memory");
    else                    asm volatile("s_wait_asynccnt 0x20" ::: "memory");
#endif
  };
  auto waitK0 = [&]() {
#if HAVE_TDM
    __builtin_amdgcn_s_wait_tensorcnt((short)0);
#else
    asm volatile("s_wait_asynccnt 0x0" ::: "memory");
#endif
  };

  v8f acc[ND];
#pragma unroll
  for (int j = 0; j < ND; ++j) acc[j] = vzero8();

  float rowM = -3.0e38f, rowL = 0.0f;   // valid in lanes 0..15 (row owners)

  Frag qf[NF];
#pragma unroll
  for (int f = 0; f < NF; ++f) {
    const __bf16* ar = Qh + (size_t)(qt * 16 + ln) * DD + f * 32 + kb;
    qf[f].q[0] = *(const u32x4*)(ar);
    qf[f].q[1] = *(const u32x4*)(ar + 16);
  }

  stageK(0, 0);                          // pipeline prologue
  int buf = 0;
  for (int kk = 0; kk < NK; kk += 32) {
    if (kk + 32 < NK) { stageK(buf ^ 1, kk + 32); waitK1(); }
    else              { waitK0(); }
    __syncthreads();                     // staged tile visible / fence

    // prefetch next V block (-> global_prefetch_b8)
    if (kk + 32 < NK)
      __builtin_prefetch(Vh + (size_t)(lane & (HD - 1)) * vtStride + kk + 32, 0, 0);

    // ---- scores for 32 keys: two 16x16 tiles, B-frags from LDS K-tile ----
    const __bf16* kt = &sK[buf][0];
    v8f s0 = vzero8(), s1 = vzero8();
#pragma unroll
    for (int f = 0; f < NF; ++f) {
      Frag b0, b1;                       // B[d, key] = K row, contiguous in d
      const __bf16* kr0 = kt + (size_t)ln * HD + f * 32 + (hi ? 8 : 0);
      const __bf16* kr1 = kt + (size_t)(16 + ln) * HD + f * 32 + (hi ? 8 : 0);
      b0.q[0] = *(const u32x4*)(kr0);
      b0.q[1] = *(const u32x4*)(kr0 + 16);
      b1.q[0] = *(const u32x4*)(kr1);
      b1.q[1] = *(const u32x4*)(kr1 + 16);
      s0 = bfwmma(qf[f].v, b0.v, s0);
      s1 = bfwmma(qf[f].v, b1.v, s1);
    }
    // C layout -> LDS [q][32 keys]
#pragma unroll
    for (int e = 0; e < 8; ++e) {
      const int r = e + 8 * hi;
      sS[r * 32 + ln]      = s0[e];
      sS[r * 32 + 16 + ln] = s1[e];
    }
    __syncthreads();
    // ---- online softmax: lanes 0..15 own one query row each ----
    if (lane < 16) {
      float mnew = rowM;
      for (int j = 0; j < 32; ++j) mnew = fmaxf(mnew, sS[lane * 32 + j] * scale);
      const float alpha = __expf(rowM - mnew);
      float ls = rowL * alpha;
      for (int j = 0; j < 32; ++j) {
        const float p = __expf(sS[lane * 32 + j] * scale - mnew);
        ls += p;
        sP[lane * 32 + j] = f2bf(p);
      }
      rowM = mnew; rowL = ls; sA[lane] = alpha;
    }
    __syncthreads();
    // ---- rescale accumulators, then O += P @ V  (EXEC uniform again) ----
    Frag pf;                               // A fragment of P[16x32]
    const __bf16* pr = &sP[ln * 32 + kb];
    pf.q[0] = *(const u32x4*)(pr);
    pf.q[1] = *(const u32x4*)(pr + 16);
#pragma unroll
    for (int j = 0; j < ND; ++j) {
#pragma unroll
      for (int e = 0; e < 8; ++e) acc[j][e] *= sA[e + 8 * hi];
      Frag vf;                             // B[key, d] from VT[d][key] contiguous
      const __bf16* vr = Vh + (size_t)(j * 16 + ln) * vtStride + kk + (hi ? 8 : 0);
      vf.q[0] = *(const u32x4*)(vr);
      vf.q[1] = *(const u32x4*)(vr + 16);
      acc[j] = bfwmma(pf.v, vf.v, acc[j]);
    }
    __syncthreads();
    buf ^= 1;
  }

  if (lane < 16) sL[lane] = rowL;
  __syncthreads();
#pragma unroll
  for (int j = 0; j < ND; ++j) {
#pragma unroll
    for (int e = 0; e < 8; ++e) {
      const int r = e + 8 * hi;
      const float o = acc[j][e] / sL[r];
      Oh[(size_t)(qt * 16 + r) * DD + j * 16 + ln] = f2bf(o);
    }
  }
}

// ---------------- residual + LayerNorm (one wave per token) ----------------
__global__ __launch_bounds__(32) void k_ln_res(const float* __restrict__ x,
                                               const float* __restrict__ y,
                                               const float* __restrict__ g,
                                               const float* __restrict__ bt,
                                               float* __restrict__ of,
                                               __bf16* __restrict__ ob) {
  const int t = blockIdx.x;
  const int lane = threadIdx.x;
  const float* xr = x + (size_t)t * DD;
  const float* yr = y + (size_t)t * DD;
  float v[8];
  float s = 0.0f;
#pragma unroll
  for (int i = 0; i < 8; ++i) { v[i] = xr[lane * 8 + i] + yr[lane * 8 + i]; s += v[i]; }
#pragma unroll
  for (int o = 16; o > 0; o >>= 1) s += __shfl_xor(s, o, 32);
  const float mean = s * (1.0f / DD);
  float vs = 0.0f;
#pragma unroll
  for (int i = 0; i < 8; ++i) { const float d = v[i] - mean; vs += d * d; }
#pragma unroll
  for (int o = 16; o > 0; o >>= 1) vs += __shfl_xor(vs, o, 32);
  const float rstd = rsqrtf(vs * (1.0f / DD) + 1e-5f);
#pragma unroll
  for (int i = 0; i < 8; ++i) {
    const int c = lane * 8 + i;
    const float o = (v[i] - mean) * rstd * g[c] + bt[c];
    of[(size_t)t * DD + c] = o;
    if (ob) ob[(size_t)t * DD + c] = f2bf(o);
  }
}

// ---------------- host driver ----------------
extern "C" void kernel_launch(void* const* d_in, const int* in_sizes, int n_in,
                              void* d_out, int out_size, void* d_ws, size_t ws_size,
                              hipStream_t stream) {
  (void)in_sizes; (void)n_in; (void)out_size; (void)ws_size;
  const float* tgt        = (const float*)d_in[0];
  const float* srcs       = (const float*)d_in[2];
  const float* pos        = (const float*)d_in[3];
  const float* in_proj_w  = (const float*)d_in[4];
  const float* in_proj_b  = (const float*)d_in[5];
  const float* out_proj_w = (const float*)d_in[6];
  const float* out_proj_b = (const float*)d_in[7];
  const float* lin_q_w    = (const float*)d_in[8];
  const float* lin_q_b    = (const float*)d_in[9];
  const float* lin_k_w    = (const float*)d_in[10];
  const float* lin_k_b    = (const float*)d_in[11];
  const float* lin_v_w    = (const float*)d_in[12];
  const float* lin_v_b    = (const float*)d_in[13];
  const float* lin_o_w    = (const float*)d_in[14];
  const float* lin_o_b    = (const float*)d_in[15];
  const float* norm1_g    = (const float*)d_in[16];
  const float* norm1_b    = (const float*)d_in[17];
  const float* norm2_g    = (const float*)d_in[18];
  const float* norm2_b    = (const float*)d_in[19];
  const float* ffn_w1     = (const float*)d_in[20];
  const float* ffn_b1     = (const float*)d_in[21];
  const float* ffn_w2     = (const float*)d_in[22];
  const float* ffn_b2     = (const float*)d_in[23];
  const float* ffn_norm_g = (const float*)d_in[24];
  const float* ffn_norm_b = (const float*)d_in[25];

  char* p = (char*)d_ws;
  auto alloc = [&](size_t bytes) -> char* {
    char* r = p;
    p += (bytes + 255) & ~(size_t)255;
    return r;
  };

  // bf16 weights
  __bf16* w_inproj  = (__bf16*)alloc((size_t)768 * DD * 2);
  __bf16* w_outproj = (__bf16*)alloc((size_t)DD * DD * 2);
  __bf16* w_lq      = (__bf16*)alloc((size_t)DD * DD * 2);
  __bf16* w_lk      = (__bf16*)alloc((size_t)DD * DD * 2);
  __bf16* w_lv      = (__bf16*)alloc((size_t)DD * DD * 2);
  __bf16* w_lo      = (__bf16*)alloc((size_t)DD * DD * 2);
  __bf16* w_f1      = (__bf16*)alloc((size_t)FFN * DD * 2);
  __bf16* w_f2      = (__bf16*)alloc((size_t)DD * FFN * 2);
  // bf16 activations
  __bf16* qk_bf     = (__bf16*)alloc((size_t)NT * DD * 2);
  __bf16* tgt_bf    = (__bf16*)alloc((size_t)NT * DD * 2);
  __bf16* s_bf      = (__bf16*)alloc((size_t)NS * DD * 2);
  __bf16* q_sa      = (__bf16*)alloc((size_t)NT * DD * 2);
  __bf16* k_sa      = (__bf16*)alloc((size_t)NT * DD * 2);
  __bf16* v_sa      = (__bf16*)alloc((size_t)NT * DD * 2);
  __bf16* vT_self   = (__bf16*)alloc((size_t)NT * DD * 2);
  __bf16* attn_out  = (__bf16*)alloc((size_t)NT * DD * 2);
  float*  sa_proj   = (float*)alloc((size_t)NT * DD * 4);
  float*  tgt1_f    = (float*)alloc((size_t)NT * DD * 4);
  __bf16* tgt1_b    = (__bf16*)alloc((size_t)NT * DD * 2);
  __bf16* query_b   = (__bf16*)alloc((size_t)NT * DD * 2);
  __bf16* key_b     = (__bf16*)alloc((size_t)NS * DD * 2);
  __bf16* value_b   = (__bf16*)alloc((size_t)NS * DD * 2);
  __bf16* vT_cross  = (__bf16*)alloc((size_t)NS * DD * 2);
  __bf16* cross_out = (__bf16*)alloc((size_t)NT * DD * 2);
  float*  tgt2_f    = (float*)alloc((size_t)NT * DD * 4);
  float*  tgt2n_f   = (float*)alloc((size_t)NT * DD * 4);
  __bf16* tgt2n_b   = (__bf16*)alloc((size_t)NT * DD * 2);
  __bf16* ffn1_b    = (__bf16*)alloc((size_t)NT * FFN * 2);
  float*  ffn2_f    = (float*)alloc((size_t)NT * DD * 4);

  auto CV = [&](const float* s, __bf16* d, int n) {
    k_cvt_bf16<<<(n + 255) / 256, 256, 0, stream>>>(s, d, n);
  };
  CV(in_proj_w, w_inproj, 768 * DD);
  CV(out_proj_w, w_outproj, DD * DD);
  CV(lin_q_w, w_lq, DD * DD);
  CV(lin_k_w, w_lk, DD * DD);
  CV(lin_v_w, w_lv, DD * DD);
  CV(lin_o_w, w_lo, DD * DD);
  CV(ffn_w1, w_f1, FFN * DD);
  CV(ffn_w2, w_f2, DD * FFN);

  k_prep_tgt<<<(NT * DD + 255) / 256, 256, 0, stream>>>(tgt, pos, qk_bf, tgt_bf, NT * DD);
  k_prep_srcs<<<((size_t)NS * DD + 255) / 256, 256, 0, stream>>>(srcs, s_bf);

  // ---- self attention ----
  dim3 g1(NT / 32, DD / 256);
  k_gemm<false, false, true><<<g1, 128, 0, stream>>>(qk_bf, w_inproj, in_proj_b,
                                                     nullptr, q_sa, NT, DD, DD);
  k_gemm<false, false, true><<<g1, 128, 0, stream>>>(qk_bf, w_inproj + 256 * DD,
                                                     in_proj_b + 256, nullptr, k_sa, NT, DD, DD);
  k_gemm<false, false, true><<<g1, 128, 0, stream>>>(tgt_bf, w_inproj + 512 * DD,
                                                     in_proj_b + 512, nullptr, v_sa, NT, DD, DD);
  k_transpose_vself<<<(NT * DD) / 256, 256, 0, stream>>>(v_sa, vT_self);

  dim3 gfs(TLc / 16, NHD, BZc);
  k_flash<DHD><<<gfs, 32, 0, stream>>>(q_sa, k_sa, vT_self, attn_out,
                                       TLc, 0.17677669529663689f,
                                       (size_t)TLc * DD, (size_t)TLc * DD,
                                       (size_t)NHD * DHD * TLc, (size_t)DHD * TLc,
                                       DHD, TLc);
  k_gemm<false, true, false><<<g1, 128, 0, stream>>>(attn_out, w_outproj, out_proj_b,
                                                     sa_proj, nullptr, NT, DD, DD);
  k_ln_res<<<NT, 32, 0, stream>>>(tgt, sa_proj, norm2_g, norm2_b, tgt1_f, tgt1_b);

  // ---- cross attention ----
  k_gemm<false, false, true><<<g1, 128, 0, stream>>>(tgt1_b, w_lq, lin_q_b,
                                                     nullptr, query_b, NT, DD, DD);
  dim3 g2(NS / 32, DD / 256);
  k_gemm<false, false, true><<<g2, 128, 0, stream>>>(s_bf, w_lk, lin_k_b,
                                                     nullptr, key_b, NS, DD, DD);
  k_gemm<false, false, true><<<g2, 128, 0, stream>>>(s_bf, w_lv, lin_v_b,
                                                     nullptr, value_b, NS, DD, DD);
  k_transpose_vcross<<<((size_t)NS * DD) / 256, 256, 0, stream>>>(value_b, vT_cross);

  dim3 gfc(TLc / 16, 1, BZc);
  k_flash<DD><<<gfc, 32, 0, stream>>>(query_b, key_b, vT_cross, cross_out,
                                      HWc, 1.0f,
                                      (size_t)TLc * DD, (size_t)HWc * DD,
                                      (size_t)DD * HWc, (size_t)0,
                                      0, HWc);
  k_gemm<false, true, false><<<g1, 128, 0, stream>>>(cross_out, w_lo, lin_o_b,
                                                     tgt2_f, nullptr, NT, DD, DD);
  k_ln_res<<<NT, 32, 0, stream>>>(tgt1_f, tgt2_f, norm1_g, norm1_b, tgt2n_f, tgt2n_b);

  // ---- FFN ----
  dim3 g3(NT / 32, FFN / 256);
  k_gemm<true, false, true><<<g3, 128, 0, stream>>>(tgt2n_b, w_f1, ffn_b1,
                                                    nullptr, ffn1_b, NT, FFN, DD);
  dim3 g4(NT / 32, DD / 256);
  k_gemm<false, true, false><<<g4, 128, 0, stream>>>(ffn1_b, w_f2, ffn_b2,
                                                     ffn2_f, nullptr, NT, DD, FFN);
  k_ln_res<<<NT, 32, 0, stream>>>(tgt2n_f, ffn2_f, ffn_norm_g, ffn_norm_b,
                                  (float*)d_out, nullptr);
}
